// MultiHeadAttention_10917806866548
// MI455X (gfx1250) — compile-verified, matrix operand order checked by reading, not run on hardware
//
#include <hip/hip_runtime.h>
#include <hip/hip_bf16.h>

// ---------------------------------------------------------------------------
// MI455X (gfx1250) multi-head attention block, bf16 WMMA everywhere.
// ---------------------------------------------------------------------------

typedef __attribute__((ext_vector_type(16))) __bf16 v16bf;
typedef __attribute__((ext_vector_type(8)))  float  v8f;

#define WMMA_BF16(A_, B_, C_) \
  __builtin_amdgcn_wmma_f32_16x16x32_bf16(false, (A_), false, (B_), (short)0, (C_), false, false)

constexpr int kB  = 2;
constexpr int kS  = 2048;
constexpr int kE  = 1024;
constexpr int kH  = 16;
constexpr int kHS = 64;
constexpr int kM  = kB * kS;       // 4096 rows in the [B*S, E] GEMMs
constexpr float kScale = 0.125f;   // 1/sqrt(64)

// 16-bit WMMA fragment: 16 bf16 = 8 dwords per lane.
union Frag {
  v16bf v;
  __bf16 h[16];
  unsigned u[8];
};

// Load a 16-bit A/B fragment for one lane from a contiguous K-major row.
// ISA layout (CDNA5 ISA): lanes 0-15 hold K={0..7,16..23} as packed pairs,
// lanes 16-31 hold K={8..15,24..31}.
__device__ inline void load_frag_row(Frag& f, const __bf16* row, bool hi) {
  const __bf16* p = row + (hi ? 8 : 0);
#pragma unroll
  for (int j = 0; j < 4; ++j) {
    f.u[j]     = *(const unsigned*)(p + 2 * j);        // K pairs {base+0..7}
    f.u[4 + j] = *(const unsigned*)(p + 16 + 2 * j);   // K pairs {base+16..23}
  }
}

// CDNA5 async memory->LDS copy (ASYNCcnt-tracked), 8 bytes per lane.
// VDST = LDS byte offset (wave-relative, = low 32 bits of generic pointer),
// VADDR = 64-bit global address.
__device__ inline void async_copy_b64(void* lds_ptr, const void* gptr) {
  unsigned lds_off = (unsigned)(size_t)lds_ptr;
  unsigned long long ga = (unsigned long long)(size_t)gptr;
  asm volatile("global_load_async_to_lds_b64 %0, %1, off"
               :: "v"(lds_off), "v"(ga) : "memory");
}
__device__ inline void wait_async0() {
  asm volatile("s_wait_asynccnt 0x0" ::: "memory");
}

// ---------------------------------------------------------------------------
// f32 -> bf16 elementwise convert
// ---------------------------------------------------------------------------
__global__ void cvt_f32_bf16_kernel(const float* __restrict__ in,
                                    __bf16* __restrict__ out, int n) {
  int i = blockIdx.x * blockDim.x + threadIdx.x;
  if (i < n) out[i] = (__bf16)in[i];
}

// ---------------------------------------------------------------------------
// Transpose+convert W[E][E] f32 (K-major rows) -> Wt[E][E] bf16 (N-major rows)
// so GEMM B-fragments become contiguous-pair loads.
// ---------------------------------------------------------------------------
__global__ __launch_bounds__(256) void transpose_cvt_kernel(
    const float* __restrict__ W, __bf16* __restrict__ Wt) {
  __shared__ float tile[32][33];
  const int tx = threadIdx.x, ty = threadIdx.y;
#pragma unroll
  for (int i = 0; i < 4; ++i) {
    int kIn = blockIdx.y * 32 + ty + i * 8;
    int nIn = blockIdx.x * 32 + tx;
    tile[ty + i * 8][tx] = W[(size_t)kIn * kE + nIn];
  }
  __syncthreads();
#pragma unroll
  for (int i = 0; i < 4; ++i) {
    int nOut = blockIdx.x * 32 + ty + i * 8;
    int kOut = blockIdx.y * 32 + tx;
    Wt[(size_t)nOut * kE + kOut] = (__bf16)tile[tx][ty + i * 8];
  }
}

// ---------------------------------------------------------------------------
// GEMM: C[M,N] = A[M,K](bf16) * Wt[N,K](bf16)^T + bias.
// Block = 256 threads = 8 waves; block tile 64x128; wave tile 32x32
// (2x2 v_wmma_f32_16x16x32_bf16 accumulators, K-step 32).
// F32OUT=false: bf16 out (projections). F32OUT=true: f32 out + residual.
// ---------------------------------------------------------------------------
template <bool F32OUT>
__global__ __launch_bounds__(256) void gemm_kernel(
    const __bf16* __restrict__ A, const __bf16* __restrict__ Wt,
    const float* __restrict__ bias, __bf16* __restrict__ outb,
    float* __restrict__ outf, const float* __restrict__ residual,
    int Kdim, int Ndim) {
  const int lane  = threadIdx.x & 31;
  const int laneq = lane & 15;
  const bool hi   = lane >= 16;
  const int wave  = threadIdx.x >> 5;
  const int mbase = blockIdx.x * 64  + (wave & 1) * 32;
  const int nbase = blockIdx.y * 128 + (wave >> 1) * 32;

  v8f acc[2][2] = {};
  for (int kk = 0; kk < Kdim; kk += 32) {
    Frag a[2], bfr[2];
#pragma unroll
    for (int t = 0; t < 2; ++t) {
      load_frag_row(a[t],   A  + (size_t)(mbase + t * 16 + laneq) * Kdim + kk, hi);
      load_frag_row(bfr[t], Wt + (size_t)(nbase + t * 16 + laneq) * Kdim + kk, hi);
    }
    if (kk + 32 < Kdim) {   // hint next K-slab into cache (global_prefetch_b8)
      __builtin_prefetch(A  + (size_t)(mbase + laneq) * Kdim + kk + 32, 0, 3);
      __builtin_prefetch(Wt + (size_t)(nbase + laneq) * Kdim + kk + 32, 0, 3);
    }
#pragma unroll
    for (int mt = 0; mt < 2; ++mt)
#pragma unroll
      for (int nt = 0; nt < 2; ++nt)
        acc[mt][nt] = WMMA_BF16(a[mt].v, bfr[nt].v, acc[mt][nt]);
  }

  // C/D layout: lane -> N (laneq), vgpr r -> M = r (+8 for hi lanes).
#pragma unroll
  for (int mt = 0; mt < 2; ++mt)
#pragma unroll
    for (int nt = 0; nt < 2; ++nt) {
      const int col = nbase + nt * 16 + laneq;
      const float bv = bias[col];
#pragma unroll
      for (int r = 0; r < 8; ++r) {
        const int row = mbase + mt * 16 + r + (hi ? 8 : 0);
        const float val = acc[mt][nt][r] + bv;
        if (F32OUT)
          outf[(size_t)row * Ndim + col] = val + residual[(size_t)row * Ndim + col];
        else
          outb[(size_t)row * Ndim + col] = (__bf16)val;
      }
    }
}

// ---------------------------------------------------------------------------
// Flash attention. Reference's reshape makes each head a contiguous
// row-major [S, hs] block at offset (b*H+h)*S*hs, for Q, K, V and ctx.
// Per workgroup: 128 threads = 4 waves, 64 queries (16/wave), one (b,h).
// Scores computed transposed (S^T = K*Q^T) so softmax stats are lane-local
// (+one shfl_xor(16)) and exp'd P tiles are *already* in B-fragment layout
// for O^T = V^T * P^T. K tile staged via CDNA5 async global->LDS copies;
// V transposed at staging through VGPRs (scatter).
// ---------------------------------------------------------------------------
__global__ __launch_bounds__(128) void attn_kernel(
    const __bf16* __restrict__ Qh, const __bf16* __restrict__ Kh,
    const __bf16* __restrict__ Vh, const int* __restrict__ pad_mask,
    __bf16* __restrict__ ctx) {
  __shared__ __bf16 Ks[32][68];  // 32 keys x 64 dims (+pad; 136B pitch, 8B-aligned)
  __shared__ __bf16 Vt[64][36];  // V^T: 64 dims x 32 keys (+pad)

  const int qb   = blockIdx.x % (kS / 64);
  const int bh   = blockIdx.x / (kS / 64);
  const int b    = bh / kH;
  const size_t base = (size_t)bh * kS * kHS;

  const int lane  = threadIdx.x & 31;
  const int laneq = lane & 15;
  const bool hi   = lane >= 16;
  const int wave  = threadIdx.x >> 5;
  const int q0    = qb * 64 + wave * 16;
  const int myq   = q0 + laneq;

  // Q^T B-fragments (dims 0-31, 32-63) — rows are contiguous in memory.
  Frag qf[2];
  load_frag_row(qf[0], Qh + base + (size_t)myq * kHS,      hi);
  load_frag_row(qf[1], Qh + base + (size_t)myq * kHS + 32, hi);

  v8f o[4] = {};                 // O^T tiles: (d 16) x (q 16), d0 = 0,16,32,48
  float m = -1e30f, l = 0.f;

  const int nblocks = (qb * 64 + 63) / 32 + 1;   // causal bound for this WG
  for (int kb = 0; kb < nblocks; ++kb) {
    const int k0 = kb * 32;
    __syncthreads();   // previous block's LDS reads done
    // Stage K tile (row-major): async global->LDS, 8B per op, 4 ops/thread.
#pragma unroll
    for (int i = threadIdx.x; i < 512; i += 128) {
      int idx = i * 4, kr = idx >> 6, d = idx & 63;
      async_copy_b64(&Ks[kr][d], Kh + base + (size_t)(k0 + kr) * kHS + d);
    }
    // Stage V transposed (scatter) — through VGPRs.
    for (int i = threadIdx.x; i < 2048; i += 128) {
      int kr = i >> 6, d = i & 63;
      Vt[d][kr] = Vh[base + (size_t)(k0 + kr) * kHS + d];
    }
    wait_async0();     // K tile landed in LDS
    __syncthreads();

    // K A-fragments: 2 key tiles x 2 dim halves.
    Frag kf[2][2];
#pragma unroll
    for (int kt = 0; kt < 2; ++kt) {
      load_frag_row(kf[kt][0], &Ks[kt * 16 + laneq][0],  hi);
      load_frag_row(kf[kt][1], &Ks[kt * 16 + laneq][32], hi);
    }
    // S^T tiles: s0 = keys k0..k0+15, s1 = keys k0+16..k0+31 (x queries).
    v8f s0 = {}, s1 = {};
    s0 = WMMA_BF16(kf[0][0].v, qf[0].v, s0);
    s0 = WMMA_BF16(kf[0][1].v, qf[1].v, s0);
    s1 = WMMA_BF16(kf[1][0].v, qf[0].v, s1);
    s1 = WMMA_BF16(kf[1][1].v, qf[1].v, s1);

    // Online softmax. Element j: key = k0 + j (+8 hi) [+16 for s1], query = myq.
    float sv0[8], sv1[8];
    float rmax = -1e30f;
#pragma unroll
    for (int j = 0; j < 8; ++j) {
      const int key0 = k0 + j + (hi ? 8 : 0);
      const int key1 = key0 + 16;
      const bool ok0 = (key0 <= myq) && (pad_mask[b * kS + key0] != 0);
      const bool ok1 = (key1 <= myq) && (pad_mask[b * kS + key1] != 0);
      sv0[j] = ok0 ? s0[j] * kScale : -1e30f;
      sv1[j] = ok1 ? s1[j] * kScale : -1e30f;
      rmax = fmaxf(rmax, fmaxf(sv0[j], sv1[j]));
    }
    rmax = fmaxf(rmax, __shfl_xor(rmax, 16, 32));  // combine lane n <-> n+16
    const float mnew  = fmaxf(m, rmax);
    const float alpha = __expf(m - mnew);
    float rsum = 0.f;
    Frag pf;  // P^T B-fragment: keys 0..31 of this block x 16 queries
#pragma unroll
    for (int j = 0; j < 8; ++j) {
      const float p0 = __expf(sv0[j] - mnew);
      const float p1 = __expf(sv1[j] - mnew);
      rsum += p0 + p1;
      pf.h[j]     = (__bf16)p0;   // C-tile elems land exactly in B-frag slots
      pf.h[8 + j] = (__bf16)p1;
    }
    rsum += __shfl_xor(rsum, 16, 32);
    l = l * alpha + rsum;
    m = mnew;
#pragma unroll
    for (int t = 0; t < 4; ++t)
#pragma unroll
      for (int j = 0; j < 8; ++j) o[t][j] *= alpha;

    // O^T += V^T * P^T : 4 d-tiles, K = 32 keys.
#pragma unroll
    for (int dt = 0; dt < 4; ++dt) {
      Frag vf;
      load_frag_row(vf, &Vt[dt * 16 + laneq][0], hi);
      o[dt] = WMMA_BF16(vf.v, pf.v, o[dt]);
    }
  }

  const float inv = 1.0f / l;
#pragma unroll
  for (int dt = 0; dt < 4; ++dt)
#pragma unroll
    for (int j = 0; j < 8; ++j) {
      const int d = dt * 16 + j + (hi ? 8 : 0);
      ctx[base + (size_t)myq * kHS + d] = (__bf16)(o[dt][j] * inv);
    }
}

// ---------------------------------------------------------------------------
// LayerNorm over E per row (matches reference: std = sqrt(mean((x-mean)^2))).
// ---------------------------------------------------------------------------
__global__ __launch_bounds__(256) void layernorm_kernel(
    const float* __restrict__ x, const float* __restrict__ gamma,
    const float* __restrict__ beta, float* __restrict__ out) {
  __shared__ float red[256];
  const int row = blockIdx.x;
  const int tid = threadIdx.x;
  const float* xr = x + (size_t)row * kE;
  float xs[4];
  float s = 0.f;
#pragma unroll
  for (int j = 0; j < 4; ++j) { xs[j] = xr[tid + j * 256]; s += xs[j]; }
  red[tid] = s; __syncthreads();
  for (int off = 128; off > 0; off >>= 1) {
    if (tid < off) red[tid] += red[tid + off];
    __syncthreads();
  }
  const float mean = red[0] * (1.0f / kE);
  __syncthreads();
  float vs = 0.f;
#pragma unroll
  for (int j = 0; j < 4; ++j) { float d = xs[j] - mean; vs += d * d; }
  red[tid] = vs; __syncthreads();
  for (int off = 128; off > 0; off >>= 1) {
    if (tid < off) red[tid] += red[tid + off];
    __syncthreads();
  }
  const float stdv = sqrtf(red[0] * (1.0f / kE));
  const float rstd = 1.0f / (stdv + 1e-8f);
#pragma unroll
  for (int j = 0; j < 4; ++j) {
    const int e = tid + j * 256;
    out[(size_t)row * kE + e] = gamma[e] * ((xs[j] - mean) * rstd) + beta[e];
  }
}

// ---------------------------------------------------------------------------
// Host-side orchestration.
// ---------------------------------------------------------------------------
extern "C" void kernel_launch(void* const* d_in, const int* in_sizes, int n_in,
                              void* d_out, int out_size, void* d_ws, size_t ws_size,
                              hipStream_t stream) {
  (void)in_sizes; (void)n_in; (void)out_size; (void)ws_size;
  const float* q     = (const float*)d_in[0];
  const float* k     = (const float*)d_in[1];
  const float* v     = (const float*)d_in[2];
  const float* Wq    = (const float*)d_in[3];
  const float* bq    = (const float*)d_in[4];
  const float* Wk    = (const float*)d_in[5];
  const float* bk    = (const float*)d_in[6];
  const float* Wv    = (const float*)d_in[7];
  const float* bv    = (const float*)d_in[8];
  const float* Wo    = (const float*)d_in[9];
  const float* bo    = (const float*)d_in[10];
  const float* gamma = (const float*)d_in[11];
  const float* beta  = (const float*)d_in[12];
  // d_in[13] = attn_mask (exact causal tril; applied analytically in-kernel)
  const int* pad_mask = (const int*)d_in[14];
  float* out = (float*)d_out;

  char* ws = (char*)d_ws;
  // Workspace layout (bytes):
  __bf16* qb16 = (__bf16*)(ws + 0);          // 8 MB each activation
  __bf16* kb16 = (__bf16*)(ws + 8388608);
  __bf16* vb16 = (__bf16*)(ws + 16777216);
  __bf16* WqT  = (__bf16*)(ws + 25165824);   // 2 MB each transposed weight
  __bf16* WkT  = (__bf16*)(ws + 27262976);
  __bf16* WvT  = (__bf16*)(ws + 29360128);
  __bf16* WoT  = (__bf16*)(ws + 31457280);
  __bf16* Qh   = (__bf16*)(ws + 33554432);   // 8 MB each projection
  __bf16* Kh   = (__bf16*)(ws + 41943040);
  __bf16* Vh   = (__bf16*)(ws + 50331648);
  __bf16* ctx  = (__bf16*)(ws + 58720256);
  float*  xbuf = (float*)(ws + 67108864);    // 16 MB pre-LN residual sum

  const int nAct = kM * kE;  // 4,194,304
  // 1) Convert activations to bf16.
  cvt_f32_bf16_kernel<<<nAct / 256, 256, 0, stream>>>(q, qb16, nAct);
  cvt_f32_bf16_kernel<<<nAct / 256, 256, 0, stream>>>(k, kb16, nAct);
  cvt_f32_bf16_kernel<<<nAct / 256, 256, 0, stream>>>(v, vb16, nAct);
  // 2) Transpose+convert weights.
  dim3 tgrid(kE / 32, kE / 32), tblock(32, 8);
  transpose_cvt_kernel<<<tgrid, tblock, 0, stream>>>(Wq, WqT);
  transpose_cvt_kernel<<<tgrid, tblock, 0, stream>>>(Wk, WkT);
  transpose_cvt_kernel<<<tgrid, tblock, 0, stream>>>(Wv, WvT);
  transpose_cvt_kernel<<<tgrid, tblock, 0, stream>>>(Wo, WoT);
  // 3) Projections (bf16 WMMA, bias fused, bf16 out).
  dim3 ggrid(kM / 64, kE / 128);
  gemm_kernel<false><<<ggrid, 256, 0, stream>>>(qb16, WqT, bq, Qh, nullptr, nullptr, kE, kE);
  gemm_kernel<false><<<ggrid, 256, 0, stream>>>(kb16, WkT, bk, Kh, nullptr, nullptr, kE, kE);
  gemm_kernel<false><<<ggrid, 256, 0, stream>>>(vb16, WvT, bv, Vh, nullptr, nullptr, kE, kE);
  // 4) Flash attention: one WG per (b, h, 64-query block).
  attn_kernel<<<kB * kH * (kS / 64), 128, 0, stream>>>(Qh, Kh, Vh, pad_mask, ctx);
  // 5) Output projection + bias + residual (f32 out).
  gemm_kernel<true><<<ggrid, 256, 0, stream>>>(ctx, WoT, bo, nullptr, xbuf, q, kE, kE);
  // 6) LayerNorm -> final output.
  layernorm_kernel<<<kM, 256, 0, stream>>>(xbuf, gamma, beta, out);
}
